// Attention_46574625358482
// MI455X (gfx1250) — matrix operand order repaired and checked by reading, not compiled, not run on hardware
//
#include <hip/hip_runtime.h>
#include <hip/hip_bf16.h>

typedef __attribute__((ext_vector_type(16))) _Float16 v16h;
typedef __attribute__((ext_vector_type(8)))  _Float16 v8h;
typedef __attribute__((ext_vector_type(8)))  float    v8f;
typedef int v4i __attribute__((vector_size(16)));   // matches builtin param type

#define DIM     2048
#define DHEAD   64
#define NHEADS  16
#define NROUTES 2
#define DHID    1024
#define BATCH   4
#define SEQ     1024
#define NKV     1024
#define LKV     1025      // nk + null
#define LPAD    1056      // padded to 33*32
#define LDSB    40        // LDS row pitch in halfs (80B: 16B-aligned, conflict-free)

#if __has_builtin(__builtin_amdgcn_global_load_async_to_lds_b128)
#define HAVE_ASYNC_LDS 1
#else
#define HAVE_ASYNC_LDS 0
#endif

// ---------------- WMMA operand loaders (CDNA5 16x16x32 f16 layouts) ----------
// A (16x32, MxK), row-major source: lane<16 -> row=lane, K {0..7, 16..23};
// lane>=16 -> row=lane-16, K {8..15, 24..31}. Two 16B loads.
__device__ __forceinline__ v16h load_a16(const _Float16* __restrict__ X, int ld,
                                         int row0, int lane, int k0) {
  const int r = row0 + (lane & 15);
  const _Float16* p = X + (long)r * ld + k0 + ((lane >> 4) << 3);
  v8h a0 = *(const v8h*)p;
  v8h a1 = *(const v8h*)(p + 16);
  v16h a;
#pragma unroll
  for (int i = 0; i < 8; ++i) { a[i] = a0[i]; a[i + 8] = a1[i]; }
  return a;
}

// B (32x16, KxN) where memory holds Bt row-major (N rows of K): lane<16 -> K
// {0..7, 16..23}; lane>=16 -> K {8..15, 24..31}. Two 16B loads.
__device__ __forceinline__ v16h load_b16(const _Float16* __restrict__ W, int ld,
                                         int col0, int lane, int k0) {
  const int c = col0 + (lane & 15);
  const _Float16* p = W + (long)c * ld + k0 + ((lane >> 4) << 3);
  v8h b0 = *(const v8h*)p;
  v8h b1 = *(const v8h*)(p + 16);
  v16h b;
#pragma unroll
  for (int i = 0; i < 8; ++i) { b[i] = b0[i]; b[i + 8] = b1[i]; }
  return b;
}

__device__ __forceinline__ v8f wmma_f16(v16h a, v16h b, v8f c) {
  return __builtin_amdgcn_wmma_f32_16x16x32_f16(false, a, false, b, (short)0, c,
                                                false, false);
}

// ---------------- async global->LDS staging (32B per thread) -----------------
__device__ __forceinline__ void stage_32B(const _Float16* gp, _Float16* lp) {
#if HAVE_ASYNC_LDS
  __builtin_amdgcn_global_load_async_to_lds_b128((v4i*)gp, (v4i*)lp, 0, 0);
  __builtin_amdgcn_global_load_async_to_lds_b128((v4i*)(gp + 8), (v4i*)(lp + 8), 0, 0);
#else
  *(v8h*)lp = *(const v8h*)gp;
  *(v8h*)(lp + 8) = *(const v8h*)(gp + 8);
#endif
}

__device__ __forceinline__ void async_drain() {
#if HAVE_ASYNC_LDS
#if __has_builtin(__builtin_amdgcn_s_wait_asynccnt)
  __builtin_amdgcn_s_wait_asynccnt(0);
#else
  asm volatile("s_wait_asynccnt 0x0" ::: "memory");
#endif
#endif
}

// ---------------- shared GEMM core: block = 128 rows x 64 cols ---------------
// 4 waves split M (32 rows each); the 64-col weight strip for the current
// 32-wide K slice is staged in LDS (async) and shared by all waves.
__device__ __forceinline__ void gemm_core(const _Float16* __restrict__ X, int ldx,
                                          int row0,
                                          const _Float16* __restrict__ W, int ldw,
                                          int col0, int K,
                                          _Float16* wlds, int tid, v8f acc[2][4]) {
  const int lane = tid & 31;
  const int sr = tid >> 1;            // staging row 0..63
  const int sh = (tid & 1) << 4;      // 0 / 16 halfs
  for (int k0 = 0; k0 < K; k0 += 32) {
    stage_32B(W + (long)(col0 + sr) * ldw + k0 + sh, wlds + sr * LDSB + sh);
    async_drain();
    __syncthreads();
    v16h a0 = load_a16(X, ldx, row0, lane, k0);
    v16h a1 = load_a16(X, ldx, row0 + 16, lane, k0);
#pragma unroll
    for (int c = 0; c < 4; ++c) {
      v16h b = load_b16(wlds, LDSB, c * 16, lane, 0);
      acc[0][c] = wmma_f16(a0, b, acc[0][c]);
      acc[1][c] = wmma_f16(a1, b, acc[1][c]);
    }
    __syncthreads();
  }
}

// ---------------- elementwise kernels ---------------------------------------
__global__ void cast_f32_to_f16(const float* __restrict__ s,
                                _Float16* __restrict__ d, int n) {
  int i = blockIdx.x * blockDim.x + threadIdx.x;
  if (i < n) d[i] = (_Float16)s[i];
}

// rms_norm: x / max(||x||,1e-12) * sqrt(DIM) * gamma, cast to f16. 1 row/block.
__global__ void rmsnorm_f16(const float* __restrict__ x,
                            const float* __restrict__ gamma,
                            _Float16* __restrict__ out) {
  __shared__ float red[8];
  const long row = blockIdx.x;
  const float* xr = x + row * DIM;
  float ss = 0.f;
  for (int i = threadIdx.x; i < DIM; i += 256) { float v = xr[i]; ss += v * v; }
#pragma unroll
  for (int m = 16; m >= 1; m >>= 1) ss += __shfl_xor(ss, m, 32);
  if ((threadIdx.x & 31) == 0) red[threadIdx.x >> 5] = ss;
  __syncthreads();
  float tot = 0.f;
#pragma unroll
  for (int w = 0; w < 8; ++w) tot += red[w];
  const float scale = 45.25483399593904f / fmaxf(sqrtf(tot), 1e-12f); // sqrt(2048)
  _Float16* orow = out + row * DIM;
  for (int i = threadIdx.x; i < DIM; i += 256)
    orow[i] = (_Float16)(xr[i] * scale * gamma[i]);
}

// null kv at j=0, zero the padded tail rows/cols of K (row-major) and Vt (d-major)
__global__ void init_kv(const float* __restrict__ null_kv,
                        _Float16* __restrict__ kh, _Float16* __restrict__ vth) {
  const int bh = blockIdx.x;            // b*16+h
  const int h  = bh & 15;
  const int d  = threadIdx.x;           // 0..63
  _Float16* kb = kh  + (long)bh * LPAD * DHEAD;
  _Float16* vb = vth + (long)bh * DHEAD * LPAD;
  kb[d]              = (_Float16)null_kv[h * DHEAD + d];                   // k null
  vb[(long)d * LPAD] = (_Float16)null_kv[NHEADS * DHEAD + h * DHEAD + d];  // v null
  for (int j = LKV; j < LPAD; ++j) {
    kb[(long)j * DHEAD + d] = (_Float16)0.f;
    vb[(long)d * LPAD + j]  = (_Float16)0.f;
  }
}

// ---------------- GEMM kernels ----------------------------------------------
__global__ void gemm_q(const _Float16* __restrict__ xh,
                       const _Float16* __restrict__ wq,
                       const float* __restrict__ sq, _Float16* __restrict__ qh) {
  __shared__ _Float16 wlds[64 * LDSB];
  const int tid = threadIdx.x, lane = tid & 31, wave = tid >> 5;
  const int row0 = blockIdx.y * 128 + wave * 32;   // over b*n = 4096
  const int col0 = blockIdx.x * 64;                // over DHID
  v8f acc[2][4] = {};
  gemm_core(xh, DIM, row0, wq, DIM, col0, DIM, wlds, tid, acc);
  const int half = lane >> 4, ln = lane & 15;
#pragma unroll
  for (int mt = 0; mt < 2; ++mt)
#pragma unroll
    for (int c = 0; c < 4; ++c) {
      const int o = col0 + 16 * c + ln, h = o >> 6, d = o & 63;
#pragma unroll
      for (int j = 0; j < 8; ++j) {
        const int row = row0 + mt * 16 + half * 8 + j;   // = b*1024+nn
        const int b = row >> 10, nn = row & 1023;
        const float v = acc[mt][c][j] * sq[row];
        qh[((long)(b * NHEADS + h) * SEQ + nn) * DHEAD + d] = (_Float16)v;
      }
    }
}

__global__ void gemm_kv(const _Float16* __restrict__ chh,
                        const _Float16* __restrict__ wkv,
                        const float* __restrict__ skv,
                        _Float16* __restrict__ kh, _Float16* __restrict__ vth) {
  __shared__ _Float16 wlds[64 * LDSB];
  const int tid = threadIdx.x, lane = tid & 31, wave = tid >> 5;
  const int br = blockIdx.z, b = br >> 1, r = br & 1;
  const int row0 = blockIdx.y * 128 + wave * 32;   // context row
  const int col0 = blockIdx.x * 64;                // over DHID (route block)
  const _Float16* X = chh + (long)br * NKV * DIM;
  const _Float16* W = wkv + (long)r * DHID * DIM;
  v8f acc[2][4] = {};
  gemm_core(X, DIM, row0, W, DIM, col0, DIM, wlds, tid, acc);
  const int half = lane >> 4, ln = lane & 15;
  const float* skvr = skv + (long)br * NKV;
#pragma unroll
  for (int mt = 0; mt < 2; ++mt)
#pragma unroll
    for (int c = 0; c < 4; ++c) {
      const int o = col0 + 16 * c + ln;
      const int hh = o >> 7, dd = o & 127, h = r * 8 + hh;
      const long base_k = (long)(b * NHEADS + h) * LPAD * DHEAD;
      const long base_v = (long)(b * NHEADS + h) * DHEAD * LPAD;
#pragma unroll
      for (int j = 0; j < 8; ++j) {
        const int jrow = row0 + mt * 16 + half * 8 + j;
        const float v = acc[mt][c][j] * skvr[jrow];
        if (dd < DHEAD) kh[base_k + (long)(jrow + 1) * DHEAD + dd] = (_Float16)v;
        else            vth[base_v + (long)(dd - DHEAD) * LPAD + (jrow + 1)] = (_Float16)v;
      }
    }
}

__global__ void gemm_o(const _Float16* __restrict__ oh,
                       const _Float16* __restrict__ wo,
                       float* __restrict__ out) {
  __shared__ _Float16 wlds[64 * LDSB];
  const int tid = threadIdx.x, lane = tid & 31, wave = tid >> 5;
  const int row0 = blockIdx.y * 128 + wave * 32;   // over b*n
  const int col0 = blockIdx.x * 64;                // over DIM
  v8f acc[2][4] = {};
  gemm_core(oh, DHID, row0, wo, DHID, col0, DHID, wlds, tid, acc);
  const int half = lane >> 4, ln = lane & 15;
#pragma unroll
  for (int mt = 0; mt < 2; ++mt)
#pragma unroll
    for (int c = 0; c < 4; ++c) {
      const int o = col0 + 16 * c + ln;
#pragma unroll
      for (int j = 0; j < 8; ++j)
        out[(long)(row0 + mt * 16 + half * 8 + j) * DIM + o] = acc[mt][c][j];
    }
}

// ---------------- flash attention: wave owns 16 query rows -------------------
__global__ void attn(const _Float16* __restrict__ qh,
                     const _Float16* __restrict__ kh,
                     const _Float16* __restrict__ vth,
                     _Float16* __restrict__ oh) {
  __shared__ _Float16 plds[4][16 * 32];
  const int lane = threadIdx.x & 31, wave = threadIdx.x >> 5;
  const int bh = blockIdx.x;
  const int q0 = blockIdx.y * 64 + wave * 16;
  const _Float16* Q = qh  + (long)bh * SEQ * DHEAD;
  const _Float16* K = kh  + (long)bh * LPAD * DHEAD;
  const _Float16* V = vth + (long)bh * DHEAD * LPAD;
  const v16h qa0 = load_a16(Q, DHEAD, q0, lane, 0);
  const v16h qa1 = load_a16(Q, DHEAD, q0, lane, 32);
  v8f acc[4] = {};
  float mrow[8], lrow[8];
#pragma unroll
  for (int j = 0; j < 8; ++j) { mrow[j] = -1e30f; lrow[j] = 0.f; }
  const int half = lane >> 4, ln = lane & 15;
  _Float16* pl = plds[wave];
  const float scale = 0.125f;  // DIM_HEAD^-0.5

  for (int j0 = 0; j0 < LPAD; j0 += 32) {
    v8f s[2];
#pragma unroll
    for (int t = 0; t < 2; ++t) {
      v8f sc = {};
      v16h b0 = load_b16(K, DHEAD, j0 + 16 * t, lane, 0);
      sc = wmma_f16(qa0, b0, sc);
      v16h b1 = load_b16(K, DHEAD, j0 + 16 * t, lane, 32);
      sc = wmma_f16(qa1, b1, sc);
      const int jc = j0 + 16 * t + ln;
      const float mask = (jc < LKV) ? 0.f : -1e30f;
#pragma unroll
      for (int j = 0; j < 8; ++j) s[t][j] = sc[j] * scale + mask;
    }
    // row max over this 32-wide stripe (rows live per 16-lane half)
    float tm[8];
#pragma unroll
    for (int j = 0; j < 8; ++j) tm[j] = fmaxf(s[0][j], s[1][j]);
#pragma unroll
    for (int m = 8; m >= 1; m >>= 1)
#pragma unroll
      for (int j = 0; j < 8; ++j) tm[j] = fmaxf(tm[j], __shfl_xor(tm[j], m, 32));
    float al[8];
#pragma unroll
    for (int j = 0; j < 8; ++j) {
      const float mn = fmaxf(mrow[j], tm[j]);
      al[j] = __expf(mrow[j] - mn);
      mrow[j] = mn;
    }
    float rs[8];
#pragma unroll
    for (int j = 0; j < 8; ++j) {
      const float p0 = __expf(s[0][j] - mrow[j]);
      const float p1 = __expf(s[1][j] - mrow[j]);
      s[0][j] = p0; s[1][j] = p1; rs[j] = p0 + p1;
    }
#pragma unroll
    for (int m = 8; m >= 1; m >>= 1)
#pragma unroll
      for (int j = 0; j < 8; ++j) rs[j] += __shfl_xor(rs[j], m, 32);
#pragma unroll
    for (int j = 0; j < 8; ++j) lrow[j] = lrow[j] * al[j] + rs[j];
#pragma unroll
    for (int c = 0; c < 4; ++c)
#pragma unroll
      for (int j = 0; j < 8; ++j) acc[c][j] *= al[j];
    // P -> LDS (f16, 16x32 row-major), reload in A layout (same-wave DS is in-order)
#pragma unroll
    for (int j = 0; j < 8; ++j) {
      const int rr = half * 8 + j;
      pl[rr * 32 + ln]      = (_Float16)s[0][j];
      pl[rr * 32 + 16 + ln] = (_Float16)s[1][j];
    }
    const v16h pa = load_a16(pl, 32, 0, lane, 0);
#pragma unroll
    for (int c = 0; c < 4; ++c) {
      v16h vb = load_b16(V, LPAD, c * 16, lane, j0);   // Vt: contiguous in j
      acc[c] = wmma_f16(pa, vb, acc[c]);
    }
  }
  const int b = bh >> 4, h = bh & 15;
#pragma unroll
  for (int c = 0; c < 4; ++c) {
    const int col = h * DHEAD + c * 16 + ln;
#pragma unroll
    for (int j = 0; j < 8; ++j) {
      const int qn = q0 + half * 8 + j;
      oh[(long)(b * SEQ + qn) * DHID + col] = (_Float16)(acc[c][j] / lrow[j]);
    }
  }
}

// ---------------- launcher ---------------------------------------------------
extern "C" void kernel_launch(void* const* d_in, const int* in_sizes, int n_in,
                              void* d_out, int out_size, void* d_ws, size_t ws_size,
                              hipStream_t stream) {
  (void)in_sizes; (void)n_in; (void)out_size; (void)ws_size;
  const float* x       = (const float*)d_in[0];
  const float* ctx     = (const float*)d_in[1];
  const float* skv     = (const float*)d_in[2];
  const float* sq      = (const float*)d_in[3];
  const float* gamma   = (const float*)d_in[4];
  const float* null_kv = (const float*)d_in[5];
  const float* Wq      = (const float*)d_in[6];
  const float* Wkv     = (const float*)d_in[7];
  const float* Wo      = (const float*)d_in[8];
  float* out = (float*)d_out;

  _Float16* p    = (_Float16*)d_ws;
  _Float16* xh   = p; p += (size_t)BATCH * SEQ * DIM;           // 4096x2048
  _Float16* chh  = p; p += (size_t)BATCH * NROUTES * NKV * DIM; // 8192x2048
  _Float16* wqh  = p; p += (size_t)DHID * DIM;
  _Float16* wkvh = p; p += (size_t)2 * DHID * DIM;
  _Float16* woh  = p; p += (size_t)DIM * DHID;
  _Float16* qh   = p; p += (size_t)BATCH * NHEADS * SEQ * DHEAD;
  _Float16* kh   = p; p += (size_t)BATCH * NHEADS * LPAD * DHEAD;
  _Float16* vth  = p; p += (size_t)BATCH * NHEADS * DHEAD * LPAD;
  _Float16* oh   = p; p += (size_t)BATCH * SEQ * DHID;

  cast_f32_to_f16<<<(DHID * DIM + 255) / 256, 256, 0, stream>>>(Wq, wqh, DHID * DIM);
  cast_f32_to_f16<<<(2 * DHID * DIM + 255) / 256, 256, 0, stream>>>(Wkv, wkvh, 2 * DHID * DIM);
  cast_f32_to_f16<<<(DIM * DHID + 255) / 256, 256, 0, stream>>>(Wo, woh, DIM * DHID);
  rmsnorm_f16<<<BATCH * SEQ, 256, 0, stream>>>(x, gamma, xh);
  rmsnorm_f16<<<BATCH * NROUTES * NKV, 256, 0, stream>>>(ctx, gamma, chh);
  init_kv<<<BATCH * NHEADS, DHEAD, 0, stream>>>(null_kv, kh, vth);
  gemm_q<<<dim3(DHID / 64, BATCH * SEQ / 128), 128, 0, stream>>>(xh, wqh, sq, qh);
  gemm_kv<<<dim3(DHID / 64, NKV / 128, BATCH * NROUTES), 128, 0, stream>>>(chh, wkvh, skv, kh, vth);
  attn<<<dim3(BATCH * NHEADS, SEQ / 64), 128, 0, stream>>>(qh, kh, vth, oh);
  gemm_o<<<dim3(DIM / 64, BATCH * SEQ / 128), 128, 0, stream>>>(oh, woh, out);
}
